// GATClassifier_23648089931783
// MI455X (gfx1250) — matrix operand order repaired and checked by reading, compiled, not run on hardware
//
#include <hip/hip_runtime.h>
#include <hip/hip_bf16.h>
#include <hip/hip_fp16.h>

typedef __attribute__((ext_vector_type(16))) _Float16 v16h;
typedef __attribute__((ext_vector_type(8)))  _Float16 v8h;
typedef __attribute__((ext_vector_type(8)))  float    v8f;
typedef __attribute__((ext_vector_type(4)))  int      v4i;

#define IN_CH 128
#define HID   512     // H1*C1
#define NHEAD 8
#define CH1   64
#define CLS   2
#define NEG_SLOPE 0.2f
#define ENC_NEG_INF 0x007FFFFFu   // fenc(-inf)
#define LDP   136     // padded LDS row pitch in halfs (272B -> 4-bank rotation, no conflicts)

// ---------- optional CDNA5 async global->LDS path (guarded, cannot break build) ----
#if defined(__HIP_DEVICE_COMPILE__) && \
    __has_builtin(__builtin_amdgcn_global_load_async_to_lds_b128) && \
    __has_builtin(__builtin_amdgcn_s_wait_asynccnt)
#define USE_ASYNC_LDS 1
#else
#define USE_ASYNC_LDS 0
#endif

static __device__ __forceinline__ void copy16_g2l(void* lds_dst, const void* gsrc) {
#if USE_ASYNC_LDS
  __builtin_amdgcn_global_load_async_to_lds_b128(
      (__attribute__((address_space(1))) v4i*)gsrc,
      (__attribute__((address_space(3))) v4i*)lds_dst, 0, 0);
#else
  *(v8h*)lds_dst = *(const v8h*)gsrc;
#endif
}
static __device__ __forceinline__ void wait_async_copies() {
#if USE_ASYNC_LDS
  __builtin_amdgcn_s_wait_asynccnt(0);
#endif
}

// ---- order-preserving float<->uint encoding for atomicMax on floats ----
static __device__ __forceinline__ unsigned fenc(float f) {
  unsigned u = __float_as_uint(f);
  return (u & 0x80000000u) ? ~u : (u | 0x80000000u);
}
static __device__ __forceinline__ float fdec(unsigned u) {
  return (u & 0x80000000u) ? __uint_as_float(u & 0x7fffffffu) : __uint_as_float(~u);
}
static __device__ __forceinline__ float lrelu(float x) { return x > 0.f ? x : NEG_SLOPE * x; }

static __device__ __forceinline__ void edge_sd(const int* ei, int E, int e, int& s, int& d) {
  if (e < E) { s = ei[e]; d = ei[E + e]; }        // edge_index rows 0/1
  else       { s = e - E; d = s; }                // appended self loop
}

// ---------------- conversions ----------------
__global__ void k_f32_to_f16(const float* __restrict__ src, _Float16* __restrict__ dst, int n) {
  int i = blockIdx.x * blockDim.x + threadIdx.x;
  if (i < n) dst[i] = (_Float16)src[i];
}

// W1 [128][512] f32 (row-major, N contiguous)  ->  WhT [512][128] f16 (K contiguous per column)
__global__ void k_w1_to_f16T(const float* __restrict__ W1, _Float16* __restrict__ WhT) {
  int i = blockIdx.x * blockDim.x + threadIdx.x;
  if (i >= IN_CH * HID) return;
  int k = i / HID, c = i % HID;
  WhT[c * IN_CH + k] = (_Float16)W1[i];
}

// ---------------- per-call state init ----------------
__global__ void k_init(float* out1, unsigned* m1, float* den1,
                       unsigned* m2, float* den2, float* out,
                       const float* __restrict__ b2, int N) {
  int i = blockIdx.x * blockDim.x + threadIdx.x;
  if (i < N * HID)   out1[i] = 0.f;
  if (i < N * NHEAD) { m1[i] = ENC_NEG_INF; den1[i] = 0.f; }
  if (i < N)         { m2[i] = ENC_NEG_INF; den2[i] = 0.f; }
  if (i < N * CLS)   out[i]  = b2[i % CLS];
}

// ---------------- GEMM1: h1 = Xh @ W  (LDS-resident WMMA f16 -> f32) ----------------
// grid = ceil(N/16) blocks, block = 256 (8 waves). Block computes 16 rows x 512 cols.
// A tile (16x128) and the WHOLE transposed W (512x128) staged in LDS via async copies.
// All fragments are ds_load_b128 pairs; K-loop fully unrolled -> 16 WMMAs per wave.
__global__ void __launch_bounds__(256)
k_gemm1_wmma(const _Float16* __restrict__ Xh, const _Float16* __restrict__ WhT,
             float* __restrict__ h1, int N) {
  extern __shared__ __align__(16) char smem[];
  _Float16* sA  = (_Float16*)smem;                                  // [16][LDP]
  _Float16* sBt = (_Float16*)(smem + 16 * LDP * sizeof(_Float16));  // [512][LDP]

  const int tid  = threadIdx.x;
  const int wave = tid >> 5;
  const int lane = tid & 31;
  const int hf   = lane >> 4;     // half-wave select
  const int l    = lane & 15;
  const int tm   = blockIdx.x;

  // stage A tile (16 x 128 halfs): each thread one 16B chunk
  {
    int r    = tid >> 4;                 // row within tile
    int rowg = tm * 16 + r; if (rowg >= N) rowg = N - 1;   // clamp; stores guarded
    int c    = (tid & 15) * 8;           // half offset within row
    copy16_g2l(sA + r * LDP + c, Xh + (size_t)rowg * IN_CH + c);
  }
  // stage all of WhT (512 cols x 128 K halfs = 128KB): 32 x 16B per thread
  #pragma unroll
  for (int i = 0; i < 32; ++i) {
    int idx = i * 256 + tid;             // 8192 chunks of 8 halfs
    int c   = idx >> 4;                  // column (16 chunks per column)
    int k8  = (idx & 15) * 8;            // K offset within column
    copy16_g2l(sBt + c * LDP + k8, WhT + (size_t)idx * 8);
  }
  wait_async_copies();
  __syncthreads();

  union Acc { v8f v; float f[8]; };
  Acc acc[4] = {};

  #pragma unroll
  for (int kc = 0; kc < 4; ++kc) {
    const int k0 = kc * 32;
    // A fragment (16x32 f16, ISA layout): two contiguous 8-half runs per lane
    union { v16h v; v8h p[2]; } a;
    a.p[0] = *(const v8h*)(sA + l * LDP + k0 + 8 * hf);
    a.p[1] = *(const v8h*)(sA + l * LDP + k0 + 16 + 8 * hf);

    #pragma unroll
    for (int t = 0; t < 4; ++t) {
      const int tn  = wave * 4 + t;      // absolute col tile (0..31)
      const int col = tn * 16 + l;
      // B fragment (32x16 f16): lane = column; two 8-half K runs matching A layout
      union { v16h v; v8h p[2]; } b;
      const _Float16* bp = sBt + col * LDP + k0 + 8 * hf;
      b.p[0] = *(const v8h*)(bp);
      b.p[1] = *(const v8h*)(bp + 16);
      acc[t].v = __builtin_amdgcn_wmma_f32_16x16x32_f16(false, a.v, false, b.v,
                                                        (short)0, acc[t].v, false, false);
    }
  }

  #pragma unroll
  for (int t = 0; t < 4; ++t) {
    const int tn = wave * 4 + t;
    #pragma unroll
    for (int r = 0; r < 8; ++r) {        // C/D: reg r -> row r+8*hf, col = l
      int m = tm * 16 + r + 8 * hf;
      if (m < N) h1[(size_t)m * HID + tn * 16 + l] = acc[t].f[r];
    }
  }
}

// ---------------- layer-1 attention scores per (node, head) ----------------
__global__ void k_scores1(const float* __restrict__ h1, const float* __restrict__ a_src,
                          const float* __restrict__ a_dst,
                          float* __restrict__ as1, float* __restrict__ ad1, int N) {
  int i = blockIdx.x * blockDim.x + threadIdx.x;
  if (i >= N * NHEAD) return;
  int n = i / NHEAD, hh = i % NHEAD;
  const float* base = h1 + (size_t)n * HID + hh * CH1;
  const float* vs = a_src + hh * CH1;
  const float* vd = a_dst + hh * CH1;
  float s = 0.f, d = 0.f;
  #pragma unroll 8
  for (int c = 0; c < CH1; ++c) { float v = base[c]; s += v * vs[c]; d += v * vd[c]; }
  as1[i] = s; ad1[i] = d;
}

// ---------------- layer-1 segment softmax: max, denom, aggregate ----------------
__global__ void k_edge_max1(const int* __restrict__ ei, int E, int Etot,
                            const float* __restrict__ as1, const float* __restrict__ ad1,
                            unsigned* __restrict__ m1) {
  int i = blockIdx.x * blockDim.x + threadIdx.x;
  if (i >= Etot * NHEAD) return;
  int e = i / NHEAD, hh = i % NHEAD;
  int s, d; edge_sd(ei, E, e, s, d);
  float ev = lrelu(as1[s * NHEAD + hh] + ad1[d * NHEAD + hh]);
  atomicMax(&m1[d * NHEAD + hh], fenc(ev));
}

__global__ void k_edge_den1(const int* __restrict__ ei, int E, int Etot,
                            const float* __restrict__ as1, const float* __restrict__ ad1,
                            const unsigned* __restrict__ m1, float* __restrict__ den1) {
  int i = blockIdx.x * blockDim.x + threadIdx.x;
  if (i >= Etot * NHEAD) return;
  int e = i / NHEAD, hh = i % NHEAD;
  int s, d; edge_sd(ei, E, e, s, d);
  int di = d * NHEAD + hh;
  float ev = lrelu(as1[s * NHEAD + hh] + ad1[di]);
  atomicAdd(&den1[di], __expf(ev - fdec(m1[di])));
}

__global__ void k_edge_agg1(const int* __restrict__ ei, int E, int Etot,
                            const float* __restrict__ as1, const float* __restrict__ ad1,
                            const unsigned* __restrict__ m1, const float* __restrict__ den1,
                            const float* __restrict__ h1, float* __restrict__ out1) {
  int i = blockIdx.x * blockDim.x + threadIdx.x;     // Etot*64 ~= 54.4M, fits int
  if (i >= Etot * CH1) return;
  int e = i / CH1, c = i % CH1;
  int s, d; edge_sd(ei, E, e, s, d);
  const float* hs = h1 + (size_t)s * HID;
  __builtin_prefetch(hs + c, 0, 1);                  // global_prefetch_b8 (L2-resident gather)
  #pragma unroll
  for (int hh = 0; hh < NHEAD; ++hh) {
    int di = d * NHEAD + hh;
    float ev = lrelu(as1[s * NHEAD + hh] + ad1[di]);
    float alpha = __expf(ev - fdec(m1[di])) / den1[di];
    atomicAdd(&out1[(size_t)d * HID + hh * CH1 + c], hs[hh * CH1 + c] * alpha);
  }
}

// ---------------- bias + ELU (in place) ----------------
__global__ void k_bias_elu(float* __restrict__ out1, const float* __restrict__ b1, int N) {
  int i = blockIdx.x * blockDim.x + threadIdx.x;
  if (i >= N * HID) return;
  float v = out1[i] + b1[i % HID];
  out1[i] = v > 0.f ? v : (__expf(v) - 1.f);
}

// ---------------- layer-2: per-node h2 = hact@W2 and scores ----------------
__global__ void k_layer2_node(const float* __restrict__ hact, const float* __restrict__ W2,
                              const float* __restrict__ a_src2, const float* __restrict__ a_dst2,
                              float* __restrict__ h2, float* __restrict__ as2,
                              float* __restrict__ ad2, int N) {
  int n = blockIdx.x * blockDim.x + threadIdx.x;
  if (n >= N) return;
  const float* row = hact + (size_t)n * HID;
  float acc0 = 0.f, acc1 = 0.f;
  #pragma unroll 4
  for (int k = 0; k < HID; ++k) {
    float v = row[k];
    acc0 += v * W2[k * CLS];
    acc1 += v * W2[k * CLS + 1];
  }
  h2[n * CLS] = acc0; h2[n * CLS + 1] = acc1;
  as2[n] = acc0 * a_src2[0] + acc1 * a_src2[1];
  ad2[n] = acc0 * a_dst2[0] + acc1 * a_dst2[1];
}

// ---------------- layer-2 segment softmax ----------------
__global__ void k_edge_max2(const int* __restrict__ ei, int E, int Etot,
                            const float* __restrict__ as2, const float* __restrict__ ad2,
                            unsigned* __restrict__ m2) {
  int e = blockIdx.x * blockDim.x + threadIdx.x;
  if (e >= Etot) return;
  int s, d; edge_sd(ei, E, e, s, d);
  atomicMax(&m2[d], fenc(lrelu(as2[s] + ad2[d])));
}

__global__ void k_edge_den2(const int* __restrict__ ei, int E, int Etot,
                            const float* __restrict__ as2, const float* __restrict__ ad2,
                            const unsigned* __restrict__ m2, float* __restrict__ den2) {
  int e = blockIdx.x * blockDim.x + threadIdx.x;
  if (e >= Etot) return;
  int s, d; edge_sd(ei, E, e, s, d);
  float ev = lrelu(as2[s] + ad2[d]);
  atomicAdd(&den2[d], __expf(ev - fdec(m2[d])));
}

__global__ void k_edge_agg2(const int* __restrict__ ei, int E, int Etot,
                            const float* __restrict__ as2, const float* __restrict__ ad2,
                            const unsigned* __restrict__ m2, const float* __restrict__ den2,
                            const float* __restrict__ h2, float* __restrict__ out) {
  int e = blockIdx.x * blockDim.x + threadIdx.x;
  if (e >= Etot) return;
  int s, d; edge_sd(ei, E, e, s, d);
  float ev = lrelu(as2[s] + ad2[d]);
  float alpha = __expf(ev - fdec(m2[d])) / den2[d];
  atomicAdd(&out[d * CLS + 0], h2[s * CLS + 0] * alpha);
  atomicAdd(&out[d * CLS + 1], h2[s * CLS + 1] * alpha);
}

extern "C" void kernel_launch(void* const* d_in, const int* in_sizes, int n_in,
                              void* d_out, int out_size, void* d_ws, size_t ws_size,
                              hipStream_t stream) {
  const float* x      = (const float*)d_in[0];
  const int*   ei     = (const int*)  d_in[1];
  const float* W1     = (const float*)d_in[2];
  const float* a_src1 = (const float*)d_in[3];
  const float* a_dst1 = (const float*)d_in[4];
  const float* b1     = (const float*)d_in[5];
  const float* W2     = (const float*)d_in[6];
  const float* a_src2 = (const float*)d_in[7];
  const float* a_dst2 = (const float*)d_in[8];
  const float* b2     = (const float*)d_in[9];
  float* out = (float*)d_out;

  const int N    = in_sizes[0] / IN_CH;
  const int E    = in_sizes[1] / 2;
  const int Etot = E + N;

  // ---- workspace layout (256B aligned slices) ----
  char* p = (char*)d_ws;
  auto alloc = [&](size_t bytes) -> void* {
    void* r = (void*)p;
    p += (bytes + 255) & ~(size_t)255;
    return r;
  };
  _Float16* Xh   = (_Float16*)alloc((size_t)N * IN_CH * sizeof(_Float16));
  _Float16* WhT  = (_Float16*)alloc((size_t)IN_CH * HID * sizeof(_Float16));
  float*    h1   = (float*)   alloc((size_t)N * HID * sizeof(float));
  float*    out1 = (float*)   alloc((size_t)N * HID * sizeof(float));
  float*    as1  = (float*)   alloc((size_t)N * NHEAD * sizeof(float));
  float*    ad1  = (float*)   alloc((size_t)N * NHEAD * sizeof(float));
  unsigned* m1   = (unsigned*)alloc((size_t)N * NHEAD * sizeof(unsigned));
  float*    den1 = (float*)   alloc((size_t)N * NHEAD * sizeof(float));
  float*    h2   = (float*)   alloc((size_t)N * CLS * sizeof(float));
  float*    as2  = (float*)   alloc((size_t)N * sizeof(float));
  float*    ad2  = (float*)   alloc((size_t)N * sizeof(float));
  unsigned* m2   = (unsigned*)alloc((size_t)N * sizeof(unsigned));
  float*    den2 = (float*)   alloc((size_t)N * sizeof(float));

  const int B = 256;
  auto blocks = [&](long long n) { return (unsigned)((n + B - 1) / B); };

  k_f32_to_f16<<<blocks((long long)N * IN_CH), B, 0, stream>>>(x, Xh, N * IN_CH);
  k_w1_to_f16T<<<blocks(IN_CH * HID), B, 0, stream>>>(W1, WhT);
  k_init<<<blocks((long long)N * HID), B, 0, stream>>>(out1, m1, den1, m2, den2, out, b2, N);

  const size_t smem = (size_t)(16 + HID) * LDP * sizeof(_Float16);  // ~140 KB (CDNA5 LDS)
  k_gemm1_wmma<<<dim3((N + 15) / 16), 256, smem, stream>>>(Xh, WhT, h1, N);

  k_scores1<<<blocks((long long)N * NHEAD), B, 0, stream>>>(h1, a_src1, a_dst1, as1, ad1, N);
  k_edge_max1<<<blocks((long long)Etot * NHEAD), B, 0, stream>>>(ei, E, Etot, as1, ad1, m1);
  k_edge_den1<<<blocks((long long)Etot * NHEAD), B, 0, stream>>>(ei, E, Etot, as1, ad1, m1, den1);
  k_edge_agg1<<<blocks((long long)Etot * CH1), B, 0, stream>>>(ei, E, Etot, as1, ad1, m1, den1, h1, out1);
  k_bias_elu<<<blocks((long long)N * HID), B, 0, stream>>>(out1, b1, N);

  k_layer2_node<<<blocks(N), B, 0, stream>>>(out1, W2, a_src2, a_dst2, h2, as2, ad2, N);
  k_edge_max2<<<blocks(Etot), B, 0, stream>>>(ei, E, Etot, as2, ad2, m2);
  k_edge_den2<<<blocks(Etot), B, 0, stream>>>(ei, E, Etot, as2, ad2, m2, den2);
  k_edge_agg2<<<blocks(Etot), B, 0, stream>>>(ei, E, Etot, as2, ad2, m2, den2, h2, out);
}
